// GroupQueryAttention_57440892616911
// MI455X (gfx1250) — matrix operand order repaired and checked by reading, not compiled, not run on hardware
//
#include <hip/hip_runtime.h>
#include <hip/hip_bf16.h>

// ---------------------------------------------------------------------------
// GQA forward for MI455X (gfx1250, wave32).  All heavy math goes through
// v_wmma_f32_16x16x32_bf16 (bf16 in, f32 accumulate).  Pipeline:
//   1) X -> bf16 ; W{q,k,v,o} -> bf16 transposed (K-contiguous B operands)
//   2) WMMA GEMM (64x64 register tile / wave, B tile staged to LDS via the
//      Tensor Data Mover, double-buffered): Q/K/V projections -> [B,H,S,64]
//   3) flash-attention (online softmax) per (b, head, 16-query) wave
//   4) WMMA GEMM: O @ Wo -> f32 output
// ---------------------------------------------------------------------------

typedef __bf16 bf16;
typedef bf16  v16bf __attribute__((ext_vector_type(16)));
typedef float v8f   __attribute__((ext_vector_type(8)));
typedef unsigned int u32;
typedef unsigned int u32x4 __attribute__((ext_vector_type(4)));
typedef int i32x4 __attribute__((ext_vector_type(4)));
typedef int i32x8 __attribute__((ext_vector_type(8)));

#define WMMA_BF16(a, b, c) \
  __builtin_amdgcn_wmma_f32_16x16x32_bf16(false, (a), false, (b), (short)0, (c), false, false)

union BPair { u32 u; bf16 h[2]; };

// A-operand fragment (16x32, M x K), contiguous k order (B uses the same):
//   lanes 0-15: M=lane, K=0..15 ; lanes 16-31: M=lane-16, K=16..31 (16 contiguous)
__device__ __forceinline__ v16bf ld_fragA(const bf16* p, int row_stride, int lane) {
  p += (size_t)(lane & 15) * row_stride;
  const int kb = (lane & 16) ? 16 : 0;
  v16bf f;
#pragma unroll
  for (int j = 0; j < 8; ++j) {
    const int kp = kb + 2 * j;
    BPair t; t.u = *reinterpret_cast<const u32*>(p + kp);
    f[2 * j] = t.h[0]; f[2 * j + 1] = t.h[1];
  }
  return f;
}

// B-operand fragment (32x16, K x N), contiguous k order (same as A):
//   lanes 0-15: N=lane, K=0..15 ; lanes 16-31: N=lane-16, K=16..31 (16 contiguous)
__device__ __forceinline__ v16bf ld_fragB(const bf16* p, int row_stride, int lane) {
  p += (size_t)(lane & 15) * row_stride + ((lane & 16) ? 16 : 0);
  v16bf f;
#pragma unroll
  for (int j = 0; j < 8; ++j) {
    BPair t; t.u = *reinterpret_cast<const u32*>(p + 2 * j);
    f[2 * j] = t.h[0]; f[2 * j + 1] = t.h[1];
  }
  return f;
}

// B-operand fragment with strided K (for V: key rows stride head_dim apart).
__device__ __forceinline__ v16bf ld_fragB_strided(const bf16* p, int k_stride, int lane) {
  p += (size_t)((lane & 16) ? 16 : 0) * k_stride;
  v16bf f;
#pragma unroll
  for (int j = 0; j < 16; ++j) f[j] = p[(size_t)j * k_stride];
  return f;
}

// ---------------------------------------------------------------------------
// Tensor Data Mover: 2-D tile load (rows x row_elems of bf16) into LDS.
// D# descriptor per CDNA5 ISA 8.3/8.4: group0 = {count, lds_addr, global_addr,
// type=2}; group1 = {data_size=2B, tensor dims, tile dims, dim0 stride}.
// clang-23 / therock-10.0 6-arg builtin form (extra SGPR group before cpol).
// ---------------------------------------------------------------------------
__device__ __forceinline__ void tdm_load_2d(u32 lds_off, const void* gptr,
                                            u32 rows, u32 row_elems,
                                            u32 row_stride_elems) {
  const unsigned long long ga = (unsigned long long)gptr;
  u32x4 g0;
  g0[0] = 1u;                                           // count=1 (valid user D#)
  g0[1] = lds_off;                                      // lds_addr (bytes)
  g0[2] = (u32)ga;                                      // global_addr[31:0]
  g0[3] = (u32)((ga >> 32) & 0x01FFFFFFu) | (2u << 30); // addr[56:32] | type=2
  const u32 td0 = row_stride_elems;                     // tensor dim0 (no OOB clip)
  const u32 td1 = rows;
  i32x8 g1;
  g1[0] = (int)(1u << 16);                              // data_size=1 (2 bytes)
  g1[1] = (int)((td0 & 0xFFFFu) << 16);                 // tensor_dim0 lo
  g1[2] = (int)(((td0 >> 16) & 0xFFFFu) | ((td1 & 0xFFFFu) << 16));
  g1[3] = (int)(((td1 >> 16) & 0xFFFFu) | ((row_elems & 0xFFFFu) << 16)); // tile_dim0
  g1[4] = (int)(rows & 0xFFFFu);                        // tile_dim1 (tile_dim2=0)
  g1[5] = (int)row_stride_elems;                        // tensor_dim0_stride[31:0]
  g1[6] = 0;
  g1[7] = 0;
  const i32x4 z4 = {};
  const i32x8 z8 = {};
  __builtin_amdgcn_tensor_load_to_lds(g0, g1, z4, z4, z8, 0);
}

// ---------------------------------------------------------------------------
// elementwise converters
// ---------------------------------------------------------------------------
__global__ __launch_bounds__(256) void k_f32_to_bf16(const float* __restrict__ src,
                                                     bf16* __restrict__ dst, int n) {
  int i = blockIdx.x * 256 + threadIdx.x;
  if (i < n) dst[i] = (bf16)src[i];
}

// Wt[n*K + k] = W[k*N + n]  (transpose + downconvert)
__global__ __launch_bounds__(256) void k_transpose_bf16(const float* __restrict__ W,
                                                        bf16* __restrict__ Wt, int K, int N) {
  int i = blockIdx.x * 256 + threadIdx.x;
  if (i >= K * N) return;
  int n = i / K, k = i % K;
  Wt[i] = (bf16)W[(size_t)k * N + n];
}

// ---------------------------------------------------------------------------
// GEMM: C[M,N] = A[M,K] (bf16 row-major) * Bt[N,K]^T (bf16, pre-transposed)
//   mode 0: C = f32 row-major [M,N]
//   mode 1: C = bf16 scattered to [B, Hx, S, 64]  (row=b*S+s, col=hx*64+d)
// 64 threads = 2 waves; each wave owns a 64x64 C tile (4 M-frags x 4 N-frags,
// 16 WMMAs per 32-deep K step).  The shared 64x32 B tile is staged into LDS
// by the Tensor Data Mover (double buffered, s_wait_tensorcnt + barrier).
// ---------------------------------------------------------------------------
__global__ __launch_bounds__(64) void k_gemm_wmma(const bf16* __restrict__ A,
                                                  const bf16* __restrict__ Bt,
                                                  void* __restrict__ Cout,
                                                  int M, int N, int K,
                                                  int mode, int Sdim, int Hx) {
  __shared__ __align__(16) bf16 Bsh[2][64 * 32];  // double-buffered B tile (8 KB)

  const int lane = threadIdx.x & 31;
  const int wave = threadIdx.x >> 5;
  const int tiles_n = N >> 6;
  const int tm = blockIdx.x / tiles_n;
  const int tn = blockIdx.x % tiles_n;
  const int m0 = tm * 128 + wave * 64;
  const int n0 = tn * 64;

  v8f acc[4][4] = {};
  const bf16* Ap = A + (size_t)m0 * K;
  const bf16* Bp = Bt + (size_t)n0 * K;

  // prologue: stage first B tile
  if (wave == 0)
    tdm_load_2d((u32)(uintptr_t)&Bsh[0][0], Bp, 64, 32, (u32)K);

  for (int k0 = 0; k0 < K; k0 += 32) {
    const int cur = (k0 >> 5) & 1;
    if (wave == 0) __builtin_amdgcn_s_wait_tensorcnt(0);  // current tile landed
    __syncthreads();
    if (wave == 0 && (k0 + 32) < K)                       // prefetch next tile
      tdm_load_2d((u32)(uintptr_t)&Bsh[cur ^ 1][0], Bp + k0 + 32, 64, 32, (u32)K);

    v16bf a[4];
#pragma unroll
    for (int mt = 0; mt < 4; ++mt) {
      __builtin_prefetch(Ap + (size_t)(mt * 16) * K + k0 + 256, 0, 0);
      a[mt] = ld_fragA(Ap + (size_t)(mt * 16) * K + k0, K, lane);
    }
#pragma unroll
    for (int c = 0; c < 4; ++c) {
      v16bf b = ld_fragB(&Bsh[cur][c * 16 * 32], 32, lane);
#pragma unroll
      for (int mt = 0; mt < 4; ++mt)
        acc[mt][c] = WMMA_BF16(a[mt], b, acc[mt][c]);
    }
    __syncthreads();  // all reads of Bsh[cur] done before TDM may rewrite it
  }

  const int hi = (lane & 16) ? 8 : 0;  // C rows: vgpr r -> m = r + hi
  const int nc = lane & 15;
  if (mode == 0) {
    float* O = (float*)Cout;
#pragma unroll
    for (int mt = 0; mt < 4; ++mt)
#pragma unroll
      for (int c = 0; c < 4; ++c)
#pragma unroll
        for (int r = 0; r < 8; ++r)
          O[(size_t)(m0 + mt * 16 + hi + r) * N + (n0 + c * 16 + nc)] = acc[mt][c][r];
  } else {
    bf16* O = (bf16*)Cout;
#pragma unroll
    for (int mt = 0; mt < 4; ++mt)
#pragma unroll
      for (int c = 0; c < 4; ++c) {
        const int col = n0 + c * 16 + nc;
        const int hh = col >> 6, d = col & 63;
#pragma unroll
        for (int r = 0; r < 8; ++r) {
          const int row = m0 + mt * 16 + hi + r;
          const int bb = row / Sdim, s = row % Sdim;
          O[(((size_t)bb * Hx + hh) * Sdim + s) * 64 + d] = (bf16)acc[mt][c][r];
        }
      }
  }
}

// ---------------------------------------------------------------------------
// Flash attention: one wave per (b, head, 16-query tile); 4 waves per block.
// Q:[B,32,S,64] K,V:[B,8,S,64] bf16 ; O:[B*S, 2048] bf16
// ---------------------------------------------------------------------------
#define SQ 2048

__global__ __launch_bounds__(128) void k_gqa_attn(const bf16* __restrict__ Q,
                                                  const bf16* __restrict__ Kc,
                                                  const bf16* __restrict__ Vc,
                                                  bf16* __restrict__ O) {
  __shared__ __align__(16) bf16 Pl[4][16 * 32];  // per-wave P tile (16q x 32k)

  const int lane = threadIdx.x & 31;
  const int wave = threadIdx.x >> 5;
  const int gtile = blockIdx.x * 4 + wave;
  const int qtiles = SQ / 16;                 // 128
  const int h  = (gtile / qtiles) & 31;
  const int b  = gtile / (qtiles * 32);
  const int q0 = (gtile % qtiles) * 16;
  const int hkv = h >> 2;                     // 32 heads -> 8 KV heads

  const bf16* Qp = Q  + ((((size_t)b * 32 + h)   * SQ) + q0) * 64;
  const bf16* Kp = Kc +  (((size_t)b * 8 + hkv)  * SQ)       * 64;
  const bf16* Vp = Vc +  (((size_t)b * 8 + hkv)  * SQ)       * 64;

  // Q fragments (d = 0..31, 32..63), kept in registers for the whole loop
  const v16bf aq0 = ld_fragA(Qp + 0,  64, lane);
  const v16bf aq1 = ld_fragA(Qp + 32, 64, lane);

  v8f oacc[4] = {};
  float rm[8], rs[8];
#pragma unroll
  for (int r = 0; r < 8; ++r) { rm[r] = -3.0e38f; rs[r] = 0.0f; }

  const float scale = 0.125f;                 // 1/sqrt(64)
  const int hi = (lane & 16) ? 8 : 0;
  const int nc = lane & 15;
  bf16* pt = &Pl[wave][0];

  for (int kb = 0; kb < SQ; kb += 32) {
    // ---- scores: two 16x16 tiles (keys kb..kb+15, kb+16..kb+31) ----
    v8f sc[2];
#pragma unroll
    for (int t = 0; t < 2; ++t) {
      v16bf bk0 = ld_fragB(Kp + (size_t)(kb + t * 16) * 64 + 0,  64, lane);
      v16bf bk1 = ld_fragB(Kp + (size_t)(kb + t * 16) * 64 + 32, 64, lane);
      v8f c = {};
      c = WMMA_BF16(aq0, bk0, c);
      c = WMMA_BF16(aq1, bk1, c);
      sc[t] = c;
    }

    // ---- online softmax: block rowmax ----
    float nm[8];
#pragma unroll
    for (int r = 0; r < 8; ++r) {
      float v0 = sc[0][r] * scale, v1 = sc[1][r] * scale;
      sc[0][r] = v0; sc[1][r] = v1;
      nm[r] = fmaxf(rm[r], fmaxf(v0, v1));
    }
#pragma unroll
    for (int m = 1; m <= 8; m <<= 1)
#pragma unroll
      for (int r = 0; r < 8; ++r) nm[r] = fmaxf(nm[r], __shfl_xor(nm[r], m, 32));

    // ---- rescale accumulators, exponentiate, partial row sums ----
    float psum[8];
#pragma unroll
    for (int r = 0; r < 8; ++r) {
      const float alpha = __expf(rm[r] - nm[r]);
      rm[r] = nm[r];
      rs[r] *= alpha;
#pragma unroll
      for (int c = 0; c < 4; ++c) oacc[c][r] *= alpha;
      const float p0 = __expf(sc[0][r] - nm[r]);
      const float p1 = __expf(sc[1][r] - nm[r]);
      sc[0][r] = p0; sc[1][r] = p1;
      psum[r] = p0 + p1;
    }
#pragma unroll
    for (int m = 1; m <= 8; m <<= 1)
#pragma unroll
      for (int r = 0; r < 8; ++r) psum[r] += __shfl_xor(psum[r], m, 32);
#pragma unroll
    for (int r = 0; r < 8; ++r) rs[r] += psum[r];

    // ---- P (C-layout f32) -> LDS bf16 16x32 -> A fragment ----
    __syncthreads();   // protect against prior iteration's ds reads
#pragma unroll
    for (int r = 0; r < 8; ++r) {
      pt[(hi + r) * 32 + nc]      = (bf16)sc[0][r];
      pt[(hi + r) * 32 + 16 + nc] = (bf16)sc[1][r];
    }
    __syncthreads();
    const v16bf ap = ld_fragA(pt, 32, lane);

    // ---- O += P * V (4 d-tiles of 16) ----
#pragma unroll
    for (int c = 0; c < 4; ++c) {
      v16bf bv = ld_fragB_strided(Vp + (size_t)kb * 64 + c * 16 + nc, 64, lane);
      oacc[c] = WMMA_BF16(ap, bv, oacc[c]);
    }
  }

  // ---- normalize and store O as [B*S, 2048] bf16 ----
#pragma unroll
  for (int r = 0; r < 8; ++r) {
    const float inv = 1.0f / rs[r];
#pragma unroll
    for (int c = 0; c < 4; ++c) {
      O[(size_t)(b * SQ + q0 + hi + r) * 2048 + (h * 64 + c * 16 + nc)] =
          (bf16)(oacc[c][r] * inv);
    }
  }
}

// ---------------------------------------------------------------------------
// host launch
// ---------------------------------------------------------------------------
extern "C" void kernel_launch(void* const* d_in, const int* in_sizes, int n_in,
                              void* d_out, int out_size, void* d_ws, size_t ws_size,
                              hipStream_t stream) {
  const float* X  = (const float*)d_in[0];
  const float* Wq = (const float*)d_in[1];
  const float* Wk = (const float*)d_in[2];
  const float* Wv = (const float*)d_in[3];
  const float* Wo = (const float*)d_in[4];
  float* out = (float*)d_out;

  const int B = 2, S = 2048, HD = 2048, H = 32, HKV = 8;
  const int M = B * S;            // 4096
  const int NKV = HKV * 64;       // 512

  char* ws = (char*)d_ws;
  bf16* Xb  = (bf16*)ws;  ws += (size_t)M * HD * 2;
  bf16* Wqt = (bf16*)ws;  ws += (size_t)HD * HD * 2;
  bf16* Wkt = (bf16*)ws;  ws += (size_t)HD * NKV * 2;
  bf16* Wvt = (bf16*)ws;  ws += (size_t)HD * NKV * 2;
  bf16* Wot = (bf16*)ws;  ws += (size_t)HD * HD * 2;
  bf16* Qb  = (bf16*)ws;  ws += (size_t)B * H  * S * 64 * 2;
  bf16* Kb  = (bf16*)ws;  ws += (size_t)B * HKV * S * 64 * 2;
  bf16* Vb  = (bf16*)ws;  ws += (size_t)B * HKV * S * 64 * 2;
  bf16* Ob  = (bf16*)ws;  ws += (size_t)M * HD * 2;

  // 1) convert / transpose
  {
    int n = M * HD;
    k_f32_to_bf16<<<(n + 255) / 256, 256, 0, stream>>>(X, Xb, n);
    n = HD * HD;
    k_transpose_bf16<<<(n + 255) / 256, 256, 0, stream>>>(Wq, Wqt, HD, HD);
    k_transpose_bf16<<<(n + 255) / 256, 256, 0, stream>>>(Wo, Wot, HD, HD);
    n = HD * NKV;
    k_transpose_bf16<<<(n + 255) / 256, 256, 0, stream>>>(Wk, Wkt, HD, NKV);
    k_transpose_bf16<<<(n + 255) / 256, 256, 0, stream>>>(Wv, Wvt, HD, NKV);
  }

  // 2) projections (QKV scatter mode): block tile 128x64, 2 waves of 64 threads
  k_gemm_wmma<<<(M / 128) * (HD  / 64), 64, 0, stream>>>(Xb, Wqt, Qb, M, HD,  HD, 1, S, H);
  k_gemm_wmma<<<(M / 128) * (NKV / 64), 64, 0, stream>>>(Xb, Wkt, Kb, M, NKV, HD, 1, S, HKV);
  k_gemm_wmma<<<(M / 128) * (NKV / 64), 64, 0, stream>>>(Xb, Wvt, Vb, M, NKV, HD, 1, S, HKV);

  // 3) attention: B*H*(S/16) = 8192 q-tiles, 4 per block
  k_gqa_attn<<<(B * H * (S / 16)) / 4, 128, 0, stream>>>(Qb, Kb, Vb, Ob);

  // 4) output projection -> f32
  k_gemm_wmma<<<(M / 128) * (HD / 64), 64, 0, stream>>>(Ob, Wot, out, M, HD, HD, 0, S, H);
}
